// CQattn_78649441124985
// MI455X (gfx1250) — matrix operand order repaired and checked by reading, not compiled, not run on hardware
//
#include <hip/hip_runtime.h>
#include <hip/hip_bf16.h>

#define NEGV (-1e30f)

typedef __attribute__((ext_vector_type(16))) __bf16        v16bf;
typedef __attribute__((ext_vector_type(8)))  float         v8f;
typedef __attribute__((ext_vector_type(8)))  unsigned int  v8u;

static __device__ __forceinline__ unsigned short f2bf(float f) {
    unsigned int u = __builtin_bit_cast(unsigned int, f);
    u += 0x7FFFu + ((u >> 16) & 1u);      // round-to-nearest-even
    return (unsigned short)(u >> 16);
}

static __device__ __forceinline__ v16bf pack16(uint4 a, uint4 b) {
    v8u t;
    t[0] = a.x; t[1] = a.y; t[2] = a.z; t[3] = a.w;
    t[4] = b.x; t[5] = b.y; t[6] = b.z; t[7] = b.w;
    return __builtin_bit_cast(v16bf, t);
}

// ---------------- r = rowwise dot(src[row,:], w) --------------------------
__global__ __launch_bounds__(256)
void rowdot_kernel(const float* __restrict__ src, const float* __restrict__ w,
                   float* __restrict__ out, int rows, int D) {
    int gwave = (blockIdx.x * blockDim.x + threadIdx.x) >> 5;
    int lane  = threadIdx.x & 31;
    if (gwave >= rows) return;
    const float4* p4 = (const float4*)(src + (size_t)gwave * D);
    const float4* w4 = (const float4*)w;
    float s = 0.f;
    for (int i = lane; i < (D >> 2); i += 32) {
        float4 a = p4[i], b = w4[i];
        s += a.x * b.x + a.y * b.y + a.z * b.z + a.w * b.w;
    }
    #pragma unroll
    for (int off = 16; off; off >>= 1) s += __shfl_xor(s, off, 32);
    if (lane == 0) out[gwave] = s;
}

// ---------------- f32 -> bf16 x4 (optionally scaled by w[d]) --------------
__global__ __launch_bounds__(256)
void conv_bf16_kernel(const float* __restrict__ src, unsigned short* __restrict__ dst,
                      const float* __restrict__ scale, int n4, int D) {
    int i = blockIdx.x * blockDim.x + threadIdx.x;   // vector index (4 floats)
    if (i >= n4) return;
    int idx = i * 4;
    float4 v = *(const float4*)(src + idx);
    if (scale) {
        v.x *= scale[(idx + 0) & (D - 1)];
        v.y *= scale[(idx + 1) & (D - 1)];
        v.z *= scale[(idx + 2) & (D - 1)];
        v.w *= scale[(idx + 3) & (D - 1)];
    }
    uint2 o;
    o.x = (unsigned int)f2bf(v.x) | ((unsigned int)f2bf(v.y) << 16);
    o.y = (unsigned int)f2bf(v.z) | ((unsigned int)f2bf(v.w) << 16);
    *(uint2*)(dst + idx) = o;
}

// ---------------- bf16 WMMA GEMM: C = op(A) * op(B) (+rank-1 bias) --------
// 256 threads = 8 waves; 128(rows) x 64(cols) tile; K step 32.
// Waves: 4 (rows) x 2 (cols), each owns a 32x32 sub-tile (2x2 WMMA frags).
// Double-buffered LDS + register staging. Per iteration:
//   ds_store tile k (waits prev loads, covered by prev iter's compute)
//   barrier                      (only dscnt must drain here)
//   issue global loads tile k+1  (waited at next iter's ds_store)
//   ds_load_b128 fragments + 4x WMMA
// A_TRANS: logical A[i][k] = Asrc[k*lda + i]   (else Asrc[i*lda + k])
// B_TRANS: logical B[k][j] = Bsrc[j*ldb + k]   (else Bsrc[k*ldb + j])
template<bool A_TRANS, bool B_TRANS, bool ADD_BIAS>
__global__ __launch_bounds__(256)
void gemm_bf16_kernel(const unsigned short* __restrict__ A,
                      const unsigned short* __restrict__ Bm,
                      float* __restrict__ Cout,
                      const float* __restrict__ biasRow,
                      const float* __restrict__ biasCol,
                      int K, int lda, int ldb, int ldc,
                      long long strideA, long long strideB, long long strideC,
                      long long strideBiasRow, long long strideBiasCol)
{
    __shared__ unsigned short At[2][128][40];  // [buf][row][k], 80B stride
    __shared__ unsigned short Bc[2][64][40];   // [buf][col][k]

    const int b    = blockIdx.z;
    const int col0 = blockIdx.x * 64;
    const int row0 = blockIdx.y * 128;

    const unsigned short* Abase = A + (size_t)b * strideA
                                + (A_TRANS ? (size_t)row0 : (size_t)row0 * lda);
    const unsigned short* Bbase = Bm + (size_t)b * strideB
                                + (B_TRANS ? (size_t)col0 * ldb : (size_t)col0);
    float* Cb = Cout + (size_t)b * strideC;

    const int tid  = threadIdx.x;
    const int wave = tid >> 5;
    const int lane = tid & 31;
    const int wr   = (wave >> 1) * 32;   // 0,32,64,96
    const int wc   = (wave & 1)  * 32;   // 0,32
    const int l15  = lane & 15;
    const int hi   = lane >> 4;

    // ---- staging address components (constant per thread) ----
    const int ar0 = tid >> 2, akc = (tid & 3) * 8;    // A non-trans
    const int ak0 = tid >> 4, ai0 = (tid & 15) * 8;   // A trans
    const int bk0 = tid >> 3, bj0 = (tid & 7) * 8;    // B non-trans
    const int bjt = tid >> 2, bkc = (tid & 3) * 8;    // B trans

    uint4 va0, va1, vb0;

    auto load_stage = [&](int kk) {
        if constexpr (!A_TRANS) {
            const unsigned short* s0 = Abase + (size_t)ar0 * lda + kk + akc;
            const unsigned short* s1 = s0 + (size_t)64 * lda;
            va0 = *(const uint4*)s0;
            va1 = *(const uint4*)s1;
            __builtin_prefetch(s0 + 64, 0, 1);
        } else {
            const unsigned short* s0 = Abase + (size_t)(kk + ak0) * lda + ai0;
            const unsigned short* s1 = s0 + (size_t)16 * lda;
            va0 = *(const uint4*)s0;
            va1 = *(const uint4*)s1;
            __builtin_prefetch(s0 + (size_t)64 * lda, 0, 1);
        }
        if constexpr (!B_TRANS) {
            const unsigned short* s0 = Bbase + (size_t)(kk + bk0) * ldb + bj0;
            vb0 = *(const uint4*)s0;
            __builtin_prefetch(s0 + (size_t)64 * ldb, 0, 1);
        } else {
            const unsigned short* s0 = Bbase + (size_t)bjt * ldb + kk + bkc;
            vb0 = *(const uint4*)s0;
            __builtin_prefetch(s0 + 64, 0, 1);
        }
    };

    auto store_stage = [&](int buf) {
        if constexpr (!A_TRANS) {
            *(uint4*)&At[buf][ar0][akc]      = va0;
            *(uint4*)&At[buf][ar0 + 64][akc] = va1;
        } else {
            unsigned int w0[4] = {va0.x, va0.y, va0.z, va0.w};
            unsigned int w1[4] = {va1.x, va1.y, va1.z, va1.w};
            #pragma unroll
            for (int q = 0; q < 8; ++q) {
                At[buf][ai0 + q][ak0]      = (unsigned short)(w0[q >> 1] >> ((q & 1) * 16));
                At[buf][ai0 + q][ak0 + 16] = (unsigned short)(w1[q >> 1] >> ((q & 1) * 16));
            }
        }
        if constexpr (!B_TRANS) {
            unsigned int w0[4] = {vb0.x, vb0.y, vb0.z, vb0.w};
            #pragma unroll
            for (int q = 0; q < 8; ++q)
                Bc[buf][bj0 + q][bk0] = (unsigned short)(w0[q >> 1] >> ((q & 1) * 16));
        } else {
            *(uint4*)&Bc[buf][bjt][bkc] = vb0;
        }
    };

    v8f acc[2][2];
    v8f z = {0.f,0.f,0.f,0.f,0.f,0.f,0.f,0.f};
    acc[0][0] = z; acc[0][1] = z; acc[1][0] = z; acc[1][1] = z;

    load_stage(0);
    int cur = 0;
    for (int kk = 0; kk < K; kk += 32) {
        store_stage(cur);                       // waits prev global loads
        __syncthreads();                        // needs only dscnt drain
        if (kk + 32 < K) load_stage(kk + 32);   // issue next tile after barrier

        // ---- fragments: two ds_load_b128 each ----
        v16bf afrag[2], bfrag[2];
        #pragma unroll
        for (int rf = 0; rf < 2; ++rf) {
            int row = wr + rf * 16 + l15;
            uint4 u0 = *(const uint4*)&At[cur][row][hi * 8];
            uint4 u1 = *(const uint4*)&At[cur][row][16 + hi * 8];
            afrag[rf] = pack16(u0, u1);
        }
        #pragma unroll
        for (int cf = 0; cf < 2; ++cf) {
            int col = wc + cf * 16 + l15;
            uint4 u0 = *(const uint4*)&Bc[cur][col][hi * 8];
            uint4 u1 = *(const uint4*)&Bc[cur][col][16 + hi * 8];
            bfrag[cf] = pack16(u0, u1);
        }

        #pragma unroll
        for (int rf = 0; rf < 2; ++rf)
            #pragma unroll
            for (int cf = 0; cf < 2; ++cf)
                acc[rf][cf] = __builtin_amdgcn_wmma_f32_16x16x32_bf16(
                    false, afrag[rf], false, bfrag[cf],
                    (short)0, acc[rf][cf], false, false);

        cur ^= 1;
    }

    // ---- store: C/D layout: row = vgpr + 8*hi, col = l15 ----
    const float* brow = ADD_BIAS ? (biasRow + (size_t)b * strideBiasRow) : nullptr;
    const float* bcol = ADD_BIAS ? (biasCol + (size_t)b * strideBiasCol) : nullptr;
    #pragma unroll
    for (int rf = 0; rf < 2; ++rf) {
        #pragma unroll
        for (int cf = 0; cf < 2; ++cf) {
            int col = col0 + wc + cf * 16 + l15;
            #pragma unroll
            for (int v = 0; v < 8; ++v) {
                int row = row0 + wr + rf * 16 + hi * 8 + v;
                float val = acc[rf][cf][v];
                if (ADD_BIAS) val += brow[row] + bcol[col];
                Cb[(size_t)row * ldc + col] = val;
            }
        }
    }
}

// ---------------- softmax over m (rows of S), Qmask -> NEG ----------------
__global__ __launch_bounds__(256)
void softmax_rows_kernel(const float* __restrict__ S, const int* __restrict__ qmask,
                         unsigned short* __restrict__ P, int N, int Mc)
{
    __shared__ float red[256];
    int row = blockIdx.x;            // b*N + n
    int b   = row / N;
    int tid = threadIdx.x;
    const float* Srow = S + (size_t)row * Mc;
    const int*   mrow = qmask + (size_t)b * Mc;

    float x0 = mrow[tid]       ? NEGV : Srow[tid];          // Mc == 512
    float x1 = mrow[tid + 256] ? NEGV : Srow[tid + 256];
    red[tid] = fmaxf(x0, x1);
    __syncthreads();
    for (int s = 128; s; s >>= 1) {
        if (tid < s) red[tid] = fmaxf(red[tid], red[tid + s]);
        __syncthreads();
    }
    float mx = red[0];
    __syncthreads();
    float e0 = __expf(x0 - mx), e1 = __expf(x1 - mx);
    red[tid] = e0 + e1;
    __syncthreads();
    for (int s = 128; s; s >>= 1) {
        if (tid < s) red[tid] += red[tid + s];
        __syncthreads();
    }
    float inv = 1.f / red[0];
    unsigned short* Prow = P + (size_t)row * Mc;
    Prow[tid]       = f2bf(e0 * inv);
    Prow[tid + 256] = f2bf(e1 * inv);
}

// ---------------- softmax over n (columns of S), Cmask -> NEG -------------
__global__ __launch_bounds__(256)
void softmax_cols_kernel(const float* __restrict__ S, const int* __restrict__ cmask,
                         unsigned short* __restrict__ P, int N, int Mc)
{
    __shared__ float red[256];
    int idx = blockIdx.x;            // b*Mc + m
    int b   = idx / Mc;
    int m   = idx - b * Mc;
    int tid = threadIdx.x;
    const float* Sb = S + (size_t)b * N * Mc;
    const int*   mb = cmask + (size_t)b * N;

    float x[8];                      // N == 2048 = 8 * 256
    float mx = -3.4e38f;
    #pragma unroll
    for (int i = 0; i < 8; ++i) {
        int n = tid + i * 256;
        float v = Sb[(size_t)n * Mc + m];
        if (mb[n]) v = NEGV;
        x[i] = v;
        mx = fmaxf(mx, v);
    }
    red[tid] = mx;
    __syncthreads();
    for (int s = 128; s; s >>= 1) {
        if (tid < s) red[tid] = fmaxf(red[tid], red[tid + s]);
        __syncthreads();
    }
    mx = red[0];
    __syncthreads();
    float sum = 0.f;
    #pragma unroll
    for (int i = 0; i < 8; ++i) { x[i] = __expf(x[i] - mx); sum += x[i]; }
    red[tid] = sum;
    __syncthreads();
    for (int s = 128; s; s >>= 1) {
        if (tid < s) red[tid] += red[tid + s];
        __syncthreads();
    }
    float inv = 1.f / red[0];
    unsigned short* Pb = P + (size_t)b * N * Mc;
    #pragma unroll
    for (int i = 0; i < 8; ++i) {
        int n = tid + i * 256;
        Pb[(size_t)n * Mc + m] = f2bf(x[i] * inv);
    }
}

extern "C" void kernel_launch(void* const* d_in, const int* in_sizes, int n_in,
                              void* d_out, int out_size, void* d_ws, size_t ws_size,
                              hipStream_t stream) {
    const float* C     = (const float*)d_in[0];
    const float* Q     = (const float*)d_in[1];
    const int*   Cmask = (const int*)d_in[2];
    const int*   Qmask = (const int*)d_in[3];
    const float* w     = (const float*)d_in[4];

    const int B = 32, N = 2048, M = 512, D = 512;

    // ---- workspace carve-up ----
    char* ws = (char*)d_ws;
    float*          r1   = (float*)ws;                               // B*N
    float*          r2   = r1 + (size_t)B * N;                       // B*M
    unsigned short* Cw3b = (unsigned short*)(r2 + (size_t)B * M);    // B*N*D bf16
    unsigned short* Qb   = Cw3b + (size_t)B * N * D;                 // B*M*D bf16
    unsigned short* Cbf  = Qb   + (size_t)B * M * D;                 // B*N*D bf16
    float*          S    = (float*)(Cbf + (size_t)B * N * D);        // B*N*M f32
    unsigned short* S1b  = (unsigned short*)(S + (size_t)B * N * M); // B*N*M bf16
    unsigned short* S2b  = S1b + (size_t)B * N * M;                  // B*N*M bf16
    float*          Tf   = (float*)(S2b + (size_t)B * N * M);        // B*M*D f32
    unsigned short* Tb   = (unsigned short*)(Tf + (size_t)B * M * D);// B*M*D bf16

    const int nC = B * N * D, nQ = B * M * D;

    // ---- bias vectors and bf16 operands ----
    rowdot_kernel<<<(B * N) / 8, 256, 0, stream>>>(C, w,     r1, B * N, D);
    rowdot_kernel<<<(B * M) / 8, 256, 0, stream>>>(Q, w + D, r2, B * M, D);
    conv_bf16_kernel<<<(nC / 4) / 256, 256, 0, stream>>>(C, Cw3b, w + 2 * D, nC / 4, D);
    conv_bf16_kernel<<<(nQ / 4) / 256, 256, 0, stream>>>(Q, Qb,   nullptr,   nQ / 4, D);
    conv_bf16_kernel<<<(nC / 4) / 256, 256, 0, stream>>>(C, Cbf,  nullptr,   nC / 4, D);

    // ---- S = (C*w3) @ Q^T + r1[:,None] + r2[None,:] ----
    dim3 gS(M / 64, N / 128, B);
    gemm_bf16_kernel<false, true, true><<<gS, 256, 0, stream>>>(
        Cw3b, Qb, S, r1, r2, D, D, D, M,
        (long long)N * D, (long long)M * D, (long long)N * M, N, M);

    // ---- masked softmaxes ----
    softmax_rows_kernel<<<B * N, 256, 0, stream>>>(S, Qmask, S1b, N, M);
    softmax_cols_kernel<<<B * M, 256, 0, stream>>>(S, Cmask, S2b, N, M);

    float* Aout = (float*)d_out;
    float* Bout = Aout + (size_t)B * N * D;

    // ---- A = S1 @ Q ----
    dim3 gA(D / 64, N / 128, B);
    gemm_bf16_kernel<false, false, false><<<gA, 256, 0, stream>>>(
        S1b, Qb, Aout, nullptr, nullptr, M, M, D, D,
        (long long)N * M, (long long)M * D, (long long)N * D, 0, 0);

    // ---- T = S2^T @ C ----
    dim3 gT(D / 64, M / 128, B);
    gemm_bf16_kernel<true, false, false><<<gT, 256, 0, stream>>>(
        S2b, Cbf, Tf, nullptr, nullptr, N, M, D, D,
        (long long)N * M, (long long)N * D, (long long)M * D, 0, 0);

    conv_bf16_kernel<<<(nQ / 4) / 256, 256, 0, stream>>>(Tf, Tb, nullptr, nQ / 4, D);

    // ---- Bout = S1 @ T ----
    gemm_bf16_kernel<false, false, false><<<gA, 256, 0, stream>>>(
        S1b, Tb, Bout, nullptr, nullptr, M, M, D, D,
        (long long)N * M, (long long)M * D, (long long)N * D, 0, 0);
}
